// MultiHeadSelfAttention_14233521619304
// MI455X (gfx1250) — matrix operand order repaired and checked by reading, compiled, not run on hardware
//
#include <hip/hip_runtime.h>
#include <stdint.h>

typedef __bf16 v16bf __attribute__((ext_vector_type(16)));
typedef float  v8f   __attribute__((ext_vector_type(8)));

#define DMODEL 1024
#define SEQ    2048
#define BATCH  2
#define NHEAD  16
#define DK     64
#define ROWS   (BATCH * SEQ)   // 4096

// ---------- CDNA5 async global->LDS copy (guarded) ----------

#if defined(__has_builtin)
#if __has_builtin(__builtin_amdgcn_global_load_async_to_lds_b128)
#define HAVE_ASYNC_LDS 1
#endif
#endif
#ifndef HAVE_ASYNC_LDS
#define HAVE_ASYNC_LDS 0
#endif

#if HAVE_ASYNC_LDS
typedef int v4i __attribute__((vector_size(16)));
typedef __attribute__((address_space(1))) v4i* as1_v4i_ptr;
typedef __attribute__((address_space(3))) v4i* as3_v4i_ptr;
#endif

// copy 16 bytes global -> LDS (per-lane)
__device__ __forceinline__ void async_cp16(const void* g, void* l) {
#if HAVE_ASYNC_LDS
  __builtin_amdgcn_global_load_async_to_lds_b128(
      (as1_v4i_ptr)g, (as3_v4i_ptr)l, 0, 0);
#else
  *(uint4*)l = *(const uint4*)g;
#endif
}

__device__ __forceinline__ void wait_async() {
#if HAVE_ASYNC_LDS
#if __has_builtin(__builtin_amdgcn_s_wait_asynccnt)
  __builtin_amdgcn_s_wait_asynccnt(0);
#else
  asm volatile("s_wait_asynccnt 0x0" ::: "memory");
#endif
#endif
}

// ---------- helpers ----------

__device__ __forceinline__ unsigned short f2bf(float f) {
  union { float f; unsigned u; } v; v.f = f;
  unsigned r = v.u + 0x7FFFu + ((v.u >> 16) & 1u);   // round-to-nearest-even
  return (unsigned short)(r >> 16);
}

// A-operand fragment (16x32 bf16, MxK): lane holds K = {kb..kb+7, kb+16..kb+23}
__device__ __forceinline__ v16bf ld_frag_a(const unsigned short* p) {
  v16bf r;
  ((uint4*)&r)[0] = *(const uint4*)(p);
  ((uint4*)&r)[1] = *(const uint4*)(p + 16);
  return r;
}
// B-operand fragment (32x16 bf16, KxN): lane holds K = {kb..kb+7, kb+16..kb+23}
__device__ __forceinline__ v16bf ld_frag_b(const unsigned short* p) {
  v16bf r;
  ((uint4*)&r)[0] = *(const uint4*)(p);
  ((uint4*)&r)[1] = *(const uint4*)(p + 16);
  return r;
}

__device__ __forceinline__ v8f wmma_bf16(v16bf a, v16bf b, v8f c) {
  return __builtin_amdgcn_wmma_f32_16x16x32_bf16(
      /*neg_a=*/false, a, /*neg_b=*/false, b,
      /*c_mod=*/(short)0, c, /*reuse_a=*/false, /*reuse_b=*/false);
}

// ---------- fp32 -> bf16 elementwise ----------

__global__ __launch_bounds__(256) void cvt_kernel(const float* __restrict__ in,
                                                  unsigned short* __restrict__ out,
                                                  int n) {
  int i = (blockIdx.x * 256 + threadIdx.x) * 4;
  if (i + 3 < n) {
    float4 v = *(const float4*)(in + i);
    unsigned short o[4] = { f2bf(v.x), f2bf(v.y), f2bf(v.z), f2bf(v.w) };
    *(uint2*)(out + i) = *(uint2*)o;
  }
}

// ---------- fp32 W[k][n] -> bf16 Wt[n][k] (tiled 32x32 transpose) ----------

__global__ __launch_bounds__(256) void cvt_w_transpose(
    const float* __restrict__ in, unsigned short* __restrict__ out) {
  __shared__ unsigned short tile[32][33];
  const int k0 = blockIdx.y * 32;
  const int n0 = blockIdx.x * 32;
  {
    int kk = threadIdx.x >> 3;
    int ns = (threadIdx.x & 7) * 4;
    float4 v = *(const float4*)(in + (size_t)(k0 + kk) * DMODEL + n0 + ns);
    tile[kk][ns + 0] = f2bf(v.x);
    tile[kk][ns + 1] = f2bf(v.y);
    tile[kk][ns + 2] = f2bf(v.z);
    tile[kk][ns + 3] = f2bf(v.w);
  }
  __syncthreads();
  {
    int n  = threadIdx.x >> 3;
    int ks = (threadIdx.x & 7) * 4;
    unsigned short o[4] = { tile[ks + 0][n], tile[ks + 1][n],
                            tile[ks + 2][n], tile[ks + 3][n] };
    *(uint2*)(out + (size_t)(n0 + n) * DMODEL + k0 + ks) = *(uint2*)o;
  }
}

// ---------- tiled bf16 WMMA GEMM:  C[4096,1024] = A * Wt^T + bias ----------
// WG tile 128x128, 8 waves in 4x2, wave tile 32x64, K-step 32, double-buffered
// MODE 0: store bf16 natural; MODE 1: bf16 per-head-transposed Vt; MODE 2: fp32

template <int MODE>
__global__ __launch_bounds__(256) void gemm_bf16(
    const unsigned short* __restrict__ A,
    const unsigned short* __restrict__ Wt,
    const float* __restrict__ bias,
    void* __restrict__ outp) {
  __shared__ unsigned short sA[2][128 * 48];   // [row][k], pitch 48
  __shared__ unsigned short sB[2][128 * 48];   // [n][k],   pitch 48

  const int tid  = threadIdx.x;
  const int lane = tid & 31;
  const int wid  = tid >> 5;
  const int half = lane >> 4;
  const int l16  = lane & 15;

  const int n0 = blockIdx.x * 128;
  const int m0 = blockIdx.y * 128;

  const int waveM = (wid >> 1) * 32;  // 0 / 32 / 64 / 96
  const int waveN = (wid & 1) * 64;   // 0 / 64

  auto stage = [&](int k0, int buf) {
#pragma unroll
    for (int j = 0; j < 2; ++j) {      // A: 128x32, 2 chunks of 16B per thread
      int c = tid * 2 + j;
      int r = c >> 2, cs = (c & 3) * 8;
      async_cp16(A + (size_t)(m0 + r) * DMODEL + k0 + cs,
                 &sA[buf][r * 48 + cs]);
    }
#pragma unroll
    for (int j = 0; j < 2; ++j) {      // B: 128x32
      int c = tid * 2 + j;
      int r = c >> 2, cs = (c & 3) * 8;
      async_cp16(Wt + (size_t)(n0 + r) * DMODEL + k0 + cs,
                 &sB[buf][r * 48 + cs]);
    }
  };

  v8f acc[2][4] = {};

  stage(0, 0);
  wait_async();
  __syncthreads();

  int cur = 0;
  for (int k0 = 0; k0 < DMODEL; k0 += 32) {
    int nxt = cur ^ 1;
    if (k0 + 32 < DMODEL) stage(k0 + 32, nxt);   // overlap with compute

    v16bf af[2], bfrag[4];
#pragma unroll
    for (int tm = 0; tm < 2; ++tm)
      af[tm] = ld_frag_a(&sA[cur][(waveM + tm * 16 + l16) * 48 + half * 8]);
#pragma unroll
    for (int tn = 0; tn < 4; ++tn)
      bfrag[tn] = ld_frag_b(&sB[cur][(waveN + tn * 16 + l16) * 48 + half * 8]);
#pragma unroll
    for (int tm = 0; tm < 2; ++tm)
#pragma unroll
      for (int tn = 0; tn < 4; ++tn)
        acc[tm][tn] = wmma_bf16(af[tm], bfrag[tn], acc[tm][tn]);

    wait_async();
    __syncthreads();
    cur = nxt;
  }

  // epilogue: bias add + store
#pragma unroll
  for (int tm = 0; tm < 2; ++tm) {
#pragma unroll
    for (int tn = 0; tn < 4; ++tn) {
      int n = n0 + waveN + tn * 16 + l16;
      float bv = bias[n];
#pragma unroll
      for (int r = 0; r < 8; ++r) {
        int m = m0 + waveM + tm * 16 + r + half * 8;
        float v = acc[tm][tn][r] + bv;
        if (MODE == 2) {
          ((float*)outp)[(size_t)m * DMODEL + n] = v;
        } else if (MODE == 1) {
          int bb = m >> 11, s = m & (SEQ - 1);
          int h = n >> 6, d = n & (DK - 1);
          ((unsigned short*)outp)[(size_t)((bb * NHEAD + h) * DK + d) * SEQ + s] =
              f2bf(v);
        } else {
          ((unsigned short*)outp)[(size_t)m * DMODEL + n] = f2bf(v);
        }
      }
    }
  }
}

// ---------- attention: one WG per (b, h, 64-query block) ----------
// 8 waves = 4 q-tiles (16 rows each) x 2 key-subsets (1024 keys each).

__global__ __launch_bounds__(256) void attn_kernel(
    const unsigned short* __restrict__ Q,    // [4096,1024] bf16
    const unsigned short* __restrict__ K,    // [4096,1024] bf16
    const unsigned short* __restrict__ Vt,   // [(b*16+h)*64+d][2048] bf16
    unsigned short* __restrict__ Ctx) {      // [4096,1024] bf16
  __shared__ unsigned short sQ[64 * 64];     // 64-row query tile (8 KB)
  __shared__ unsigned short sP[8 * 512];     // per-wave P staging (16x32 each)
  __shared__ float sM[8 * 16];
  __shared__ float sL[8 * 16];
  __shared__ float fM[64];
  __shared__ float fLi[64];
  __shared__ float sCtx[64 * 64];            // 16 KB f32 reduction buffer

  const int tid  = threadIdx.x;
  const int lane = tid & 31;
  const int wid  = tid >> 5;
  const int half = lane >> 4;
  const int l16  = lane & 15;
  const int tq   = wid & 3;   // which 16-row q-tile
  const int ks   = wid >> 2;  // which key subset (0/1)

  const int bid = blockIdx.x;
  const int b   = bid >> 9;               // 16*32 blocks per batch
  const int h   = (bid >> 5) & (NHEAD - 1);
  const int qb  = bid & 31;               // 64-row block index

  const float SCALE = 0.125f;  // 1/sqrt(64)

  // zero context reduction buffer (4096 floats)
#pragma unroll
  for (int j = 0; j < 16; ++j) sCtx[tid * 16 + j] = 0.f;

  // stage Q tile (64x64 bf16) via async copy: 512 16B chunks
#pragma unroll
  for (int j = 0; j < 2; ++j) {
    int c = tid * 2 + j;
    int r = c >> 3, cs = (c & 7) * 8;
    async_cp16(Q + (size_t)(b * SEQ + qb * 64 + r) * DMODEL + h * DK + cs,
               &sQ[r * 64 + cs]);
  }
  wait_async();
  __syncthreads();

  // constant Q A-fragments for this wave's q-tile
  v16bf aq[2];
#pragma unroll
  for (int c = 0; c < 2; ++c)
    aq[c] = ld_frag_a(&sQ[(tq * 16 + l16) * 64 + c * 32 + half * 8]);

  const unsigned short* Kbase = K + (size_t)(b * SEQ) * DMODEL + h * DK;

  // score tile for a 16-key block: S = Q K^T (raw, unscaled)
  auto score_tile = [&](int kb16) -> v8f {
    v8f s = {};
    const unsigned short* kp = Kbase + (size_t)(kb16 * 16 + l16) * DMODEL;
#pragma unroll
    for (int c = 0; c < 2; ++c) {
      v16bf bk = ld_frag_b(kp + c * 32 + half * 8);
      s = wmma_bf16(aq[c], bk, s);
    }
    return s;
  };

  // ---- pass 1: LANE-LOCAL online (m,l) over this wave's 64 key-blocks ----
  float mrun[8], lrun[8];
#pragma unroll
  for (int r = 0; r < 8; ++r) { mrun[r] = -1e30f; lrun[r] = 0.f; }

  for (int it = 0; it < 64; ++it) {
    int kb16 = ks + it * 2;
    if (it + 1 < 64)   // WGP-scope prefetch of next block's K row
      __builtin_prefetch(Kbase + (size_t)((kb16 + 2) * 16 + l16) * DMODEL, 0, 3);
    v8f s = score_tile(kb16);
#pragma unroll
    for (int r = 0; r < 8; ++r) {
      float sv = s[r] * SCALE;
      float nm = fmaxf(mrun[r], sv);
      lrun[r] = lrun[r] * __expf(mrun[r] - nm) + __expf(sv - nm);
      mrun[r] = nm;
    }
  }
  // merge (m,l) across the 16 lanes owning each row (one-time shuffle tree)
#pragma unroll
  for (int r = 0; r < 8; ++r) {
    float m = mrun[r], l = lrun[r];
#pragma unroll
    for (int msk = 1; msk < 16; msk <<= 1) {
      float om = __shfl_xor(m, msk);
      float ol = __shfl_xor(l, msk);
      float nm = fmaxf(m, om);
      l = l * __expf(m - nm) + ol * __expf(om - nm);
      m = nm;
    }
    mrun[r] = m; lrun[r] = l;
  }
  if (l16 == 0) {
#pragma unroll
    for (int r = 0; r < 8; ++r) {
      int row = r + half * 8;
      sM[wid * 16 + row] = mrun[r];
      sL[wid * 16 + row] = lrun[r];
    }
  }
  __syncthreads();
  // combine the two key-subset waves for each (q-tile, row)
  if (tid < 64) {
    int t = tid >> 4, row = tid & 15;
    float m0 = sM[t * 16 + row],      l0 = sL[t * 16 + row];
    float m1 = sM[(t + 4) * 16 + row], l1 = sL[(t + 4) * 16 + row];
    float M = fmaxf(m0, m1);
    float L = l0 * __expf(m0 - M) + l1 * __expf(m1 - M);
    fM[tid]  = M;
    fLi[tid] = 1.0f / L;
  }
  __syncthreads();

  float mr[8], li[8];
#pragma unroll
  for (int r = 0; r < 8; ++r) {
    int row = tq * 16 + r + half * 8;
    mr[r] = fM[row];
    li[r] = fLi[row];
  }

  // ---- pass 2: P = softmax(S); context += P * V ----
  v8f acc[4] = {};
  const unsigned short* Vbase = Vt + (size_t)((b * NHEAD + h) * DK) * SEQ;

  for (int it = 0; it < 32; ++it) {
    int chunk = ks + it * 2;  // 32-key chunk handled by this wave
#pragma unroll
    for (int sub = 0; sub < 2; ++sub) {
      v8f s = score_tile(chunk * 2 + sub);
#pragma unroll
      for (int r = 0; r < 8; ++r) {
        float p = __expf(s[r] * SCALE - mr[r]) * li[r];
        // C-layout -> LDS (row-major 16x32 per wave) for A-operand reload
        sP[wid * 512 + (r + half * 8) * 32 + sub * 16 + l16] = f2bf(p);
      }
    }
    __syncthreads();
    v16bf pa = ld_frag_a(&sP[wid * 512 + l16 * 32 + half * 8]);
#pragma unroll
    for (int tn = 0; tn < 4; ++tn) {
      const unsigned short* vp =
          Vbase + (size_t)(tn * 16 + l16) * SEQ + chunk * 32 + half * 8;
      v16bf bv = ld_frag_b(vp);
      acc[tn] = wmma_bf16(pa, bv, acc[tn]);
    }
    __syncthreads();
  }

  // reduce the two key-subset waves per q-tile via LDS float atomics
#pragma unroll
  for (int tn = 0; tn < 4; ++tn)
#pragma unroll
    for (int r = 0; r < 8; ++r)
      atomicAdd(&sCtx[(tq * 16 + r + half * 8) * 64 + tn * 16 + l16], acc[tn][r]);
  __syncthreads();

  // write context tile (bf16, natural layout for the output GEMM)
#pragma unroll
  for (int j = 0; j < 16; ++j) {
    int idx = tid * 16 + j;
    int row = idx >> 6, d = idx & (DK - 1);
    Ctx[(size_t)(b * SEQ + qb * 64 + row) * DMODEL + h * DK + d] = f2bf(sCtx[idx]);
  }
}

// ---------- launch ----------

extern "C" void kernel_launch(void* const* d_in, const int* in_sizes, int n_in,
                              void* d_out, int out_size, void* d_ws, size_t ws_size,
                              hipStream_t stream) {
  const float* x  = (const float*)d_in[0];
  const float* Wq = (const float*)d_in[1];
  const float* bq = (const float*)d_in[2];
  const float* Wk = (const float*)d_in[3];
  const float* bk = (const float*)d_in[4];
  const float* Wv = (const float*)d_in[5];
  const float* bv = (const float*)d_in[6];
  const float* Wo = (const float*)d_in[7];
  const float* bo = (const float*)d_in[8];

  char* ws = (char*)d_ws;
  const size_t MB = 1024 * 1024;
  unsigned short* xb   = (unsigned short*)(ws + 0 * MB);   // x bf16        (8 MB)
  unsigned short* wqtb = (unsigned short*)(ws + 8 * MB);   // Wq^T bf16     (2 MB)
  unsigned short* wktb = (unsigned short*)(ws + 10 * MB);
  unsigned short* wvtb = (unsigned short*)(ws + 12 * MB);
  unsigned short* wotb = (unsigned short*)(ws + 14 * MB);
  unsigned short* Qb   = (unsigned short*)(ws + 16 * MB);  // Q bf16        (8 MB)
  unsigned short* Kb   = (unsigned short*)(ws + 24 * MB);  // K bf16        (8 MB)
  unsigned short* Vtb  = (unsigned short*)(ws + 32 * MB);  // V^T bf16      (8 MB)
  unsigned short* Cxb  = (unsigned short*)(ws + 40 * MB);  // context bf16  (8 MB)

  const int NX = ROWS * DMODEL;  // 4 Mi elements
  cvt_kernel<<<NX / 1024, 256, 0, stream>>>(x, xb, NX);

  dim3 gt(DMODEL / 32, DMODEL / 32);
  cvt_w_transpose<<<gt, 256, 0, stream>>>(Wq, wqtb);
  cvt_w_transpose<<<gt, 256, 0, stream>>>(Wk, wktb);
  cvt_w_transpose<<<gt, 256, 0, stream>>>(Wv, wvtb);
  cvt_w_transpose<<<gt, 256, 0, stream>>>(Wo, wotb);

  dim3 gg(DMODEL / 128, ROWS / 128);
  gemm_bf16<0><<<gg, 256, 0, stream>>>(xb, wqtb, bq, Qb);
  gemm_bf16<0><<<gg, 256, 0, stream>>>(xb, wktb, bk, Kb);
  gemm_bf16<1><<<gg, 256, 0, stream>>>(xb, wvtb, bv, Vtb);

  attn_kernel<<<BATCH * NHEAD * (SEQ / 64), 256, 0, stream>>>(Qb, Kb, Vtb, Cxb);

  gemm_bf16<2><<<gg, 256, 0, stream>>>(Cxb, wotb, bo, d_out);
}
